// BlocksCore_32349693673918
// MI455X (gfx1250) — matrix operand order repaired and checked by reading, not compiled, not run on hardware
//
#include <hip/hip_runtime.h>
#include <hip/hip_bf16.h>
#include <math.h>

typedef __attribute__((ext_vector_type(16))) _Float16 v16h;
typedef __attribute__((ext_vector_type(8)))  _Float16 v8h;
typedef __attribute__((ext_vector_type(8)))  float    v8f;

#define DEV __device__ __forceinline__

static constexpr int    BN    = 16384;   // batch
static constexpr int    NINP  = 768;
static constexpr int    NHID  = 360;
static constexpr int    NBLK  = 6;
static constexpr int    BS    = 60;

// ---------------- workspace layout (bytes), lifetime-aliased ----------------
static constexpr size_t SZ_INPH  = (size_t)BN * 768 * 2;   // inp as f16
static constexpr size_t SZ_KV    = (size_t)BN * 512 * 4;   // K(256)|V(240)+pad, f32
static constexpr size_t SZ_QG    = (size_t)BN * 1536 * 4;  // Q [B,6,256] then gates [B,6,256]
static constexpr size_t OFF_INPH = 0;
static constexpr size_t OFF_KV   = OFF_INPH + SZ_INPH;
static constexpr size_t OFF_QG   = OFF_KV + SZ_KV;
static constexpr size_t REGION0  = OFF_QG + SZ_QG;          // 159.4 MB
static constexpr size_t OFF_C    = 0;                        // comm qkv [B,6,384] f32 aliases region0
static constexpr size_t SZ_C     = (size_t)BN * 2304 * 4;    // 151 MB < REGION0  (QG dead by then)
static constexpr size_t OFF_X    = REGION0;                  // LSTM GEMM input [B,6,320] f16
static constexpr size_t SZ_X     = (size_t)BN * 1920 * 2;
static constexpr size_t OFF_HXP  = OFF_X + SZ_X;             // hx padded f16 [B,6,64]
static constexpr size_t SZ_HXP   = (size_t)BN * 384 * 2;
static constexpr size_t OFF_NUL  = OFF_HXP + SZ_HXP;         // null attn [B,6] f32
static constexpr size_t SZ_NUL   = (size_t)BN * 6 * 4;
static constexpr size_t OFF_MSK  = OFF_NUL + SZ_NUL;         // mask [B,6] f32
static constexpr size_t SZ_MSK   = (size_t)BN * 6 * 4;
static constexpr size_t OFF_HXN  = OFF_MSK + SZ_MSK;         // hx_new f32 [B,360]
static constexpr size_t SZ_HXN   = (size_t)BN * 360 * 4;
static constexpr size_t OFF_CXN  = OFF_HXN + SZ_HXN;         // cx_new f32 [B,360]
static constexpr size_t SZ_CXN   = (size_t)BN * 360 * 4;
static constexpr size_t OFF_HNP  = OFF_CXN + SZ_CXN;         // hx_new padded f16 [B,6,64]
static constexpr size_t SZ_HNP   = (size_t)BN * 384 * 2;
static constexpr size_t OFF_RAW  = OFF_HNP + SZ_HNP;         // comm attn out f16 [B*6,128]
static constexpr size_t SZ_RAW   = (size_t)BN * 6 * 128 * 2;
static constexpr size_t OFF_FG   = OFF_RAW + SZ_RAW;         // fc|gate out f32 [B*6,128]
static constexpr size_t SZ_FG    = (size_t)BN * 6 * 128 * 4;
// f16 transposed/padded weights
static constexpr size_t OFF_WKV  = OFF_FG + SZ_FG;           // [512,768]
static constexpr size_t SZ_WKV   = (size_t)512 * 768 * 2;
static constexpr size_t OFF_WQ   = OFF_WKV + SZ_WKV;         // [6][256,64]
static constexpr size_t SZ_WQ    = (size_t)6 * 256 * 64 * 2;
static constexpr size_t OFF_WG   = OFF_WQ + SZ_WQ;           // [6][256,320]
static constexpr size_t SZ_WG    = (size_t)6 * 256 * 320 * 2;
static constexpr size_t OFF_WC   = OFF_WG + SZ_WG;           // [6][384,64]
static constexpr size_t SZ_WC    = (size_t)6 * 384 * 64 * 2;
static constexpr size_t OFF_WFG  = OFF_WC + SZ_WC;           // [128,128]

// ------------------------- WMMA fragment helpers ---------------------------
// A row-major [M,lda] f16; frag rows m0..m0+15, cols k0..k0+31 (ISA 16-bit A 16x32 layout)
DEV v16h load_a_frag(const _Float16* A, int lda, int m0, int k0, int lane) {
  int m  = m0 + (lane & 15);
  int kb = (lane & 16) ? 8 : 0;
  const _Float16* p = A + (size_t)m * lda + k0 + kb;
  v8h lo = *(const v8h*)(p);
  v8h hi = *(const v8h*)(p + 16);
  v16h r;
#pragma unroll
  for (int i = 0; i < 8; ++i) { r[i] = lo[i]; r[i + 8] = hi[i]; }
  return r;
}

// B stored transposed: Wt [N,K] row-major f16; frag k0..k0+31, n0..n0+15
DEV v16h load_b_frag(const _Float16* Wt, int ldw, int n0, int k0, int lane) {
  int n  = n0 + (lane & 15);
  int kb = (lane & 16) ? 8 : 0;
  const _Float16* p = Wt + (size_t)n * ldw + k0 + kb;
  v8h lo = *(const v8h*)(p);
  v8h hi = *(const v8h*)(p + 16);
  v16h r;
#pragma unroll
  for (int i = 0; i < 8; ++i) { r[i] = lo[i]; r[i + 8] = hi[i]; }
  return r;
}

DEV void store_c_frag(float* C, int ldc, int m0, int n0, v8f acc, int lane) {
  int n  = n0 + (lane & 15);
  int mb = (lane & 16) ? 8 : 0;
#pragma unroll
  for (int r = 0; r < 8; ++r) C[(size_t)(m0 + mb + r) * ldc + n] = acc[r];
}

// Generic tiled GEMM: C[M,N] f32 = A[M,K] f16 @ Wt[N,K]^T f16.
// 8 waves/block; wave = 16 rows x 64 cols. grid(M/128, N/64, nblk); z offsets in elements.
__global__ __launch_bounds__(256) void gemm_wmma(
    const _Float16* __restrict__ A, int lda, int aZOff,
    const _Float16* __restrict__ Wt, int K, int wZOff,
    float* __restrict__ C, int ldc, int cZOff)
{
  int z = blockIdx.z;
  A  += (size_t)z * aZOff;
  Wt += (size_t)z * wZOff;
  C  += (size_t)z * cZOff;
  int lane = threadIdx.x & 31;
  int wave = threadIdx.x >> 5;
  int m0 = blockIdx.x * 128 + wave * 16;
  int n0 = blockIdx.y * 64;
  v8f a0 = {}, a1 = {}, a2 = {}, a3 = {};
  for (int k0 = 0; k0 < K; k0 += 32) {
    v16h af = load_a_frag(A, lda, m0, k0, lane);
    v16h b0 = load_b_frag(Wt, K, n0 +  0, k0, lane);
    v16h b1 = load_b_frag(Wt, K, n0 + 16, k0, lane);
    v16h b2 = load_b_frag(Wt, K, n0 + 32, k0, lane);
    v16h b3 = load_b_frag(Wt, K, n0 + 48, k0, lane);
    a0 = __builtin_amdgcn_wmma_f32_16x16x32_f16(false, af, false, b0, (short)0, a0, false, false);
    a1 = __builtin_amdgcn_wmma_f32_16x16x32_f16(false, af, false, b1, (short)0, a1, false, false);
    a2 = __builtin_amdgcn_wmma_f32_16x16x32_f16(false, af, false, b2, (short)0, a2, false, false);
    a3 = __builtin_amdgcn_wmma_f32_16x16x32_f16(false, af, false, b3, (short)0, a3, false, false);
  }
  store_c_frag(C, ldc, m0, n0 +  0, a0, lane);
  store_c_frag(C, ldc, m0, n0 + 16, a1, lane);
  store_c_frag(C, ldc, m0, n0 + 32, a2, lane);
  store_c_frag(C, ldc, m0, n0 + 48, a3, lane);
}

// ------------------------- weight prep (f32 -> f16, transpose+pad) ---------
__global__ void k_prep_kv(const float* __restrict__ Wk, const float* __restrict__ Wv,
                          _Float16* __restrict__ Wt) {
  int i = blockIdx.x * blockDim.x + threadIdx.x;
  if (i >= 512 * 768) return;
  int n = i / 768, k = i % 768;
  float v = 0.f;
  if (n < 256)       v = Wk[(size_t)(768 + k) * 256 + n];          // module 1
  else if (n < 496)  v = Wv[(size_t)(768 + k) * 240 + (n - 256)];
  Wt[(size_t)n * 768 + k] = (_Float16)v;
}

__global__ void k_prep_q(const float* __restrict__ Wq, _Float16* __restrict__ Wt) {
  int i = blockIdx.x * blockDim.x + threadIdx.x;
  if (i >= 6 * 256 * 64) return;
  int kb = i / (256 * 64), n = (i / 64) % 256, kk = i % 64;
  float v = (kk < 60) ? Wq[((size_t)kb * 60 + kk) * 256 + n] : 0.f;
  Wt[(size_t)kb * 256 * 64 + (size_t)n * 64 + kk] = (_Float16)v;
}

__global__ void k_prep_g(const float* __restrict__ Wih, const float* __restrict__ Whh,
                         _Float16* __restrict__ Wt) {
  int i = blockIdx.x * blockDim.x + threadIdx.x;
  if (i >= 6 * 256 * 320) return;
  int kb = i / (256 * 320), n = (i / 320) % 256, k = i % 320;
  float v = 0.f;
  if (n < 240) {
    int g = n / 60, j = n % 60;
    int row = g * 360 + kb * 60 + j;
    if (k < 240)                 v = Wih[(size_t)row * 1440 + kb * 240 + k];
    else if (k >= 256 && k < 316) v = Whh[(size_t)row * 360 + kb * 60 + (k - 256)];
  }
  Wt[(size_t)kb * 256 * 320 + (size_t)n * 320 + k] = (_Float16)v;
}

__global__ void k_prep_c(const float* __restrict__ Wq, const float* __restrict__ Wk,
                         const float* __restrict__ Wv, _Float16* __restrict__ Wt) {
  int i = blockIdx.x * blockDim.x + threadIdx.x;
  if (i >= 6 * 384 * 64) return;
  int kb = i / (384 * 64), n = (i / 64) % 384, kk = i % 64;
  float v = 0.f;
  if (kk < 60) {
    size_t r = (size_t)kb * 60 + kk;
    if (n < 128)      v = Wq[r * 128 + n];
    else if (n < 256) v = Wk[r * 128 + (n - 128)];
    else              v = Wv[r * 128 + (n - 256)];
  }
  Wt[(size_t)kb * 384 * 64 + (size_t)n * 64 + kk] = (_Float16)v;
}

__global__ void k_prep_fg(const float* __restrict__ fc, const float* __restrict__ gt,
                          _Float16* __restrict__ Wt) {
  int i = blockIdx.x * blockDim.x + threadIdx.x;
  if (i >= 128 * 128) return;
  int n = i / 128, k = i % 128;
  float v = 0.f;
  if (n < 60)                 v = fc[(size_t)n * 128 + k];
  else if (n >= 64 && n < 124) v = gt[(size_t)(n - 64) * 128 + k];
  Wt[i] = (_Float16)v;
}

// ------------------------- activation converts -----------------------------
__global__ void k_conv_inp(const float* __restrict__ x, _Float16* __restrict__ y, int n) {
  int i = blockIdx.x * blockDim.x + threadIdx.x;
  if (i < n) y[i] = (_Float16)x[i];
}

__global__ void k_conv_hx(const float* __restrict__ hx, _Float16* __restrict__ y) {
  int i = blockIdx.x * blockDim.x + threadIdx.x;
  if (i >= BN * 384) return;
  int b = i / 384, r = i % 384, kb = r / 64, j = r % 64;
  y[i] = (j < 60) ? (_Float16)hx[(size_t)b * 360 + kb * 60 + j] : (_Float16)0.f;
}

// -------- input attention: 2-key softmax (null key has score 0, value 0) ---
DEV float sigf(float x) { return 1.f / (1.f + expf(-x)); }

__global__ void k_attn_in(const float* __restrict__ Q, const float* __restrict__ KV,
                          const float* __restrict__ hx,
                          _Float16* __restrict__ X, float* __restrict__ nul) {
  int i = blockIdx.x * blockDim.x + threadIdx.x;
  if (i >= BN * 6) return;
  int b = i / 6, kb = i % 6;
  const float* q  = Q  + (size_t)b * 1536 + kb * 256;
  const float* kv = KV + (size_t)b * 512;
  _Float16* x = X + (size_t)b * 1920 + kb * 320;
  float nw = 0.f;
#pragma unroll
  for (int h = 0; h < 4; ++h) {
    float s = 0.f;
    for (int d = 0; d < 64; ++d) s += q[h * 64 + d] * kv[h * 64 + d];
    float w1 = sigf(s * 0.125f);
    nw += (1.f - w1) * 0.25f;
    for (int d = 0; d < 60; ++d) x[h * 60 + d] = (_Float16)(w1 * kv[256 + h * 60 + d]);
  }
  for (int d = 240; d < 256; ++d) x[d] = (_Float16)0.f;
  for (int d = 0; d < 60; ++d)    x[256 + d] = (_Float16)hx[(size_t)b * 360 + kb * 60 + d];
  x[316] = x[317] = x[318] = x[319] = (_Float16)0.f;
  nul[i] = nw;
}

// top-2 null-attention blocks get mask 0 (ties -> lower index, matching lax.top_k)
__global__ void k_mask(const float* __restrict__ nul, float* __restrict__ mask) {
  int b = blockIdx.x * blockDim.x + threadIdx.x;
  if (b >= BN) return;
  float v[6];
  for (int k = 0; k < 6; ++k) v[k] = nul[(size_t)b * 6 + k];
  int i1 = 0;
  for (int k = 1; k < 6; ++k) if (v[k] > v[i1]) i1 = k;
  int i2 = (i1 == 0) ? 1 : 0;
  for (int k = 0; k < 6; ++k) if (k != i1 && v[k] > v[i2]) i2 = k;
  for (int k = 0; k < 6; ++k) mask[(size_t)b * 6 + k] = (k == i1 || k == i2) ? 0.f : 1.f;
}

// ------------------------- LSTM elementwise --------------------------------
__global__ void k_lstm(const float* __restrict__ G, const float* __restrict__ cx,
                       const float* __restrict__ b_ih, const float* __restrict__ b_hh,
                       float* __restrict__ hxn, float* __restrict__ cxn,
                       _Float16* __restrict__ hnp) {
  int i = blockIdx.x * blockDim.x + threadIdx.x;
  if (i >= BN * 384) return;
  int b = i / 384, r = i % 384, kb = r / 64, j = r % 64;
  if (j >= 60) { hnp[i] = (_Float16)0.f; return; }
  int col = kb * 60 + j;
  const float* g = G + (size_t)b * 1536 + kb * 256;
  float gi = g[j]       + b_ih[col]        + b_hh[col];
  float gf = g[60 + j]  + b_ih[360 + col]  + b_hh[360 + col];
  float gg = g[120 + j] + b_ih[720 + col]  + b_hh[720 + col];
  float go = g[180 + j] + b_ih[1080 + col] + b_hh[1080 + col];
  float c  = sigf(gf) * cx[(size_t)b * 360 + col] + sigf(gi) * tanhf(gg);
  float hn = sigf(go) * tanhf(c);
  cxn[(size_t)b * 360 + col] = c;
  hxn[(size_t)b * 360 + col] = hn;
  hnp[i] = (_Float16)hn;
}

// -------- communication attention: 4 heads x 6 keys softmax ----------------
__global__ void k_comm(const float* __restrict__ C, _Float16* __restrict__ raw) {
  int i = blockIdx.x * blockDim.x + threadIdx.x;
  if (i >= BN * 6) return;
  int b = i / 6, kq = i % 6;
  const float* base = C + (size_t)b * 2304;
  const float* q = base + kq * 384;
#pragma unroll
  for (int h = 0; h < 4; ++h) {
    float sc[6];
    for (int kk = 0; kk < 6; ++kk) {
      float s = 0.f;
      const float* kp = base + kk * 384 + 128 + h * 32;
      for (int d = 0; d < 32; ++d) s += q[h * 32 + d] * kp[d];
      sc[kk] = s * 0.17677669529663689f;   // 1/sqrt(32)
    }
    float mx = sc[0];
    for (int kk = 1; kk < 6; ++kk) mx = fmaxf(mx, sc[kk]);
    float ss = 0.f;
    for (int kk = 0; kk < 6; ++kk) { sc[kk] = expf(sc[kk] - mx); ss += sc[kk]; }
    float inv = 1.f / ss;
    for (int d = 0; d < 32; ++d) {
      float o = 0.f;
      for (int kk = 0; kk < 6; ++kk) o += sc[kk] * base[kk * 384 + 256 + h * 32 + d];
      raw[(size_t)i * 128 + h * 32 + d] = (_Float16)(o * inv);
    }
  }
}

// ------------------------- final merge -------------------------------------
__global__ void k_final(const float* __restrict__ FG, const float* __restrict__ fc_b,
                        const float* __restrict__ gate_b, const float* __restrict__ mask,
                        const float* __restrict__ hxn, const float* __restrict__ cxn,
                        const float* __restrict__ hx, const float* __restrict__ cx,
                        float* __restrict__ outHx, float* __restrict__ outCx) {
  int i = blockIdx.x * blockDim.x + threadIdx.x;
  if (i >= BN * 360) return;
  int b = i / 360, col = i % 360, kb = col / 60, j = col % 60;
  const float* fg = FG + ((size_t)b * 6 + kb) * 128;
  float attc = sigf(fg[64 + j] + gate_b[j]) * tanhf(fg[j] + fc_b[j]);
  float hn = hxn[i] + attc;
  float m = mask[(size_t)b * 6 + kb];
  outHx[i] = m * hn     + (1.f - m) * hx[i];
  outCx[i] = m * cxn[i] + (1.f - m) * cx[i];
}

// ---------------------------------------------------------------------------
extern "C" void kernel_launch(void* const* d_in, const int* in_sizes, int n_in,
                              void* d_out, int out_size, void* d_ws, size_t ws_size,
                              hipStream_t stream) {
  const float* inp    = (const float*)d_in[0];
  const float* hx     = (const float*)d_in[1];   // [1,B,360] == [B,360]
  const float* cx     = (const float*)d_in[2];
  const float* Wq_inp = (const float*)d_in[3];
  const float* Wk_inp = (const float*)d_in[4];
  const float* Wv_inp = (const float*)d_in[5];
  const float* Wq_c   = (const float*)d_in[6];
  const float* Wk_c   = (const float*)d_in[7];
  const float* Wv_c   = (const float*)d_in[8];
  const float* fc_w   = (const float*)d_in[9];
  const float* fc_b   = (const float*)d_in[10];
  const float* gate_w = (const float*)d_in[11];
  const float* gate_b = (const float*)d_in[12];
  const float* W_ih   = (const float*)d_in[13];
  const float* W_hh   = (const float*)d_in[14];
  const float* b_ih   = (const float*)d_in[15];
  const float* b_hh   = (const float*)d_in[16];

  char* ws = (char*)d_ws;
  _Float16* inpH  = (_Float16*)(ws + OFF_INPH);
  float*    wsKV  = (float*)   (ws + OFF_KV);
  float*    wsQG  = (float*)   (ws + OFF_QG);
  float*    wsC   = (float*)   (ws + OFF_C);
  _Float16* wsX   = (_Float16*)(ws + OFF_X);
  _Float16* hxP   = (_Float16*)(ws + OFF_HXP);
  float*    nul   = (float*)   (ws + OFF_NUL);
  float*    mask  = (float*)   (ws + OFF_MSK);
  float*    hxn   = (float*)   (ws + OFF_HXN);
  float*    cxn   = (float*)   (ws + OFF_CXN);
  _Float16* hnp   = (_Float16*)(ws + OFF_HNP);
  _Float16* rawH  = (_Float16*)(ws + OFF_RAW);
  float*    wsFG  = (float*)   (ws + OFF_FG);
  _Float16* hWkv  = (_Float16*)(ws + OFF_WKV);
  _Float16* hWq   = (_Float16*)(ws + OFF_WQ);
  _Float16* hWg   = (_Float16*)(ws + OFF_WG);
  _Float16* hWc   = (_Float16*)(ws + OFF_WC);
  _Float16* hWfg  = (_Float16*)(ws + OFF_WFG);

  float* outHx = (float*)d_out;
  float* outCx = outHx + (size_t)BN * 360;

  // weight prep (tiny)
  k_prep_kv<<<1536, 256, 0, stream>>>(Wk_inp, Wv_inp, hWkv);
  k_prep_q <<< 384, 256, 0, stream>>>(Wq_inp, hWq);
  k_prep_g <<<1920, 256, 0, stream>>>(W_ih, W_hh, hWg);
  k_prep_c <<< 576, 256, 0, stream>>>(Wq_c, Wk_c, Wv_c, hWc);
  k_prep_fg<<<  64, 256, 0, stream>>>(fc_w, gate_w, hWfg);

  // activation converts
  k_conv_inp<<<BN * 768 / 256, 256, 0, stream>>>(inp, inpH, BN * 768);
  k_conv_hx <<<BN * 384 / 256, 256, 0, stream>>>(hx, hxP);

  // G1: [B,768] @ [768,512] -> K|V
  gemm_wmma<<<dim3(BN / 128, 512 / 64, 1), 256, 0, stream>>>(
      inpH, 768, 0, hWkv, 768, 0, wsKV, 512, 0);
  // G2: per block: hx[B,64] @ [64,256] -> Q
  gemm_wmma<<<dim3(BN / 128, 256 / 64, 6), 256, 0, stream>>>(
      hxP, 384, 64, hWq, 64, 256 * 64, wsQG, 1536, 256);
  // input attention + build X = [att(240)|pad|hx(60)|pad]
  k_attn_in<<<BN * 6 / 256, 256, 0, stream>>>(wsQG, wsKV, hx, wsX, nul);
  k_mask<<<BN / 256, 256, 0, stream>>>(nul, mask);
  // G3: per block: X[B,320] @ [320,256] -> gates (240 used)
  gemm_wmma<<<dim3(BN / 128, 256 / 64, 6), 256, 0, stream>>>(
      wsX, 1920, 320, hWg, 320, 256 * 320, wsQG, 1536, 256);
  // LSTM cell
  k_lstm<<<BN * 384 / 256, 256, 0, stream>>>(wsQG, cx, b_ih, b_hh, hxn, cxn, hnp);
  // G4: per block: hx_new[B,64] @ [64,384] -> q|k|v comm
  gemm_wmma<<<dim3(BN / 128, 384 / 64, 6), 256, 0, stream>>>(
      hnp, 384, 64, hWc, 64, 384 * 64, wsC, 2304, 384);
  // communication attention
  k_comm<<<BN * 6 / 256, 256, 0, stream>>>(wsC, rawH);
  // G5: [B*6,128] @ [128,128] -> fc|gate
  gemm_wmma<<<dim3(BN * 6 / 128, 128 / 64, 1), 256, 0, stream>>>(
      rawH, 128, 0, hWfg, 128, 0, wsFG, 128, 0);
  // final merge
  k_final<<<BN * 360 / 256, 256, 0, stream>>>(wsFG, fc_b, gate_b, mask, hxn, cxn,
                                              hx, cx, outHx, outCx);
}
